// WindowAttention_1812476199056
// MI455X (gfx1250) — matrix operand order repaired and checked by reading, not compiled, not run on hardware
//
#include <hip/hip_runtime.h>

typedef __attribute__((ext_vector_type(16))) _Float16 v16h;
typedef __attribute__((ext_vector_type(8)))  _Float16 v8h;
typedef __attribute__((ext_vector_type(8)))  float    v8f;

#define HEADS 6
#define HD    32
#define TBL   961          // (2*16-1)*(2*16-1)

// exchange with xor-16 partner lane (group-of-32 swizzle: and=0x1f, xor=0x10)
#define SWZ_XOR16(v) \
  __int_as_float(__builtin_amdgcn_ds_swizzle(__float_as_int(v), ((16 << 10) | 0x1f)))

__device__ __forceinline__ float bcast_lane(float v, int idx4) {
    return __int_as_float(__builtin_amdgcn_ds_bpermute(idx4, __float_as_int(v)));
}

union U16 { v16h v; v8h p[2]; _Float16 h[16]; };
union U8h { v8h  v; _Float16 h[8];  };

__global__ __launch_bounds__(256, 1)
void cswin_attn_kernel(const float* __restrict__ qkv,
                       const float* __restrict__ bias_table,
                       float* __restrict__ out)
{
    // block = (window, head); 512 windows * 6 heads = 3072 blocks
    const int head = blockIdx.x % HEADS;
    const int widx = blockIdx.x / HEADS;   // 0..511
    const int b    = widx >> 8;
    const int wy   = (widx >> 4) & 15;
    const int wx   = widx & 15;

    const int tid   = threadIdx.x;
    const int lane  = tid & 31;
    const int wave  = tid >> 5;
    const int hl    = lane & 15;
    const bool hi   = lane >= 16;
    const int rbase = hi ? 8 : 0;          // C-layout row offset for upper half

    __shared__ _Float16 Ks[256][40];       // K token-major, 20480 B
    __shared__ _Float16 Vt[32][264];       // V transposed d-major, 16896 B
    __shared__ _Float16 Ps[8][32][40];     // per-wave P scratch, 20480 B
    __shared__ float    Bth[TBL + 7];      // this head's bias slice, 3872 B

    const size_t plane = (size_t)2 * 65536 * 192;
    const float* qg = qkv;
    const float* kg = qkv + plane;
    const float* vg = qkv + 2 * plane;

    auto goff = [&](int t) -> size_t {     // global float offset of token t, chan 0 of head
        int hh = (wy << 4) + (t >> 4);
        int ww = (wx << 4) + (t & 15);
        return ((size_t)b * 65536 + (size_t)hh * 256 + ww) * 192 + head * HD;
    };

    // ---- stage K, V^T, bias into LDS ----
    {
        const float* kp = kg + goff(tid);
        const float* vp = vg + goff(tid);
        #pragma unroll
        for (int d4 = 0; d4 < 8; ++d4) {
            float4 kq = ((const float4*)kp)[d4];
            Ks[tid][4*d4+0] = (_Float16)kq.x;
            Ks[tid][4*d4+1] = (_Float16)kq.y;
            Ks[tid][4*d4+2] = (_Float16)kq.z;
            Ks[tid][4*d4+3] = (_Float16)kq.w;
            float4 vq = ((const float4*)vp)[d4];
            Vt[4*d4+0][tid] = (_Float16)vq.x;
            Vt[4*d4+1][tid] = (_Float16)vq.y;
            Vt[4*d4+2][tid] = (_Float16)vq.z;
            Vt[4*d4+3][tid] = (_Float16)vq.w;
        }
        for (int i = tid; i < TBL; i += 256)
            Bth[i] = bias_table[i * HEADS + head];
    }
    __syncthreads();

    // ---- Q^T as WMMA B-operand (per-lane contiguous d-range), pre-scaled ----
    const float SCALE = 0.17677669529663687f;
    const int q0 = wave * 32;
    U16 qb[2];
    #pragma unroll
    for (int qt = 0; qt < 2; ++qt) {
        const float* qp = qg + goff(q0 + qt * 16 + hl);
        const int dd = hi ? 16 : 0;
        #pragma unroll
        for (int j = 0; j < 16; ++j)
            qb[qt].h[j] = (_Float16)(qp[dd + j] * SCALE);
    }

    // all-ones B operand for row-sum WMMA (l = P * 1)
    U16 ones;
    #pragma unroll
    for (int j = 0; j < 16; ++j) ones.h[j] = (_Float16)1.0f;

    // bpermute byte-indices for redistributing lane-stats to C-layout rows
    int idx4[8];
    #pragma unroll
    for (int r = 0; r < 8; ++r) idx4[r] = (rbase + r) << 2;

    // strength-reduced bias base index (decreases by 62 per kb, 31 per kt)
    //   bias(n, m) = Bth[(ih-kh+15)*31 + (iw-kw+15)],  kw = rbase + r
    int cb[2];
    #pragma unroll
    for (int qt = 0; qt < 2; ++qt)
        cb[qt] = (((q0 >> 4) + qt) + 15) * 31 + (hl + 15 - rbase);

    // ---- state: running max per lane (query), O and l accumulators ----
    float mQ[2] = { -1e30f, -1e30f };
    v8f Oacc[2][2], lacc[2];
    #pragma unroll
    for (int qt = 0; qt < 2; ++qt)
        #pragma unroll
        for (int r = 0; r < 8; ++r) {
            Oacc[qt][0][r] = 0.f; Oacc[qt][1][r] = 0.f; lacc[qt][r] = 0.f;
        }

    // ---- main loop over 8 key blocks of 32 ----
    #pragma unroll 2
    for (int kb = 0; kb < 8; ++kb) {
        const int k0 = kb * 32;

        // A operand: K tiles (M=key, Kdim=d) in A-layout, from LDS
        U16 ka[2];
        #pragma unroll
        for (int kt = 0; kt < 2; ++kt) {
            const int krow = k0 + kt * 16 + hl;
            const int d0   = hi ? 16 : 0;
            ka[kt].p[0] = *(const v8h*)&Ks[krow][d0];
            ka[kt].p[1] = *(const v8h*)&Ks[krow][d0 + 8];
        }

        // S^T tiles with bias folded in as the WMMA C operand
        v8f ST[2][2];   // [kt][qt]
        #pragma unroll
        for (int kt = 0; kt < 2; ++kt)
            #pragma unroll
            for (int qt = 0; qt < 2; ++qt) {
                v8f bc;
                #pragma unroll
                for (int r = 0; r < 8; ++r)
                    bc[r] = Bth[cb[qt] - kt * 31 - r];
                ST[kt][qt] = __builtin_amdgcn_wmma_f32_16x16x32_f16(
                    false, ka[kt].v, false, qb[qt].v, (short)0, bc, false, false);
            }
        #pragma unroll
        for (int qt = 0; qt < 2; ++qt) cb[qt] -= 62;

        float alphaQ[2];
        #pragma unroll
        for (int qt = 0; qt < 2; ++qt) {
            // pairwise max over this lane's 16 keys, then xor-16 combine
            float m01[8];
            #pragma unroll
            for (int r = 0; r < 8; ++r) m01[r] = fmaxf(ST[0][qt][r], ST[1][qt][r]);
            #pragma unroll
            for (int s = 4; s > 0; s >>= 1)
                #pragma unroll
                for (int r = 0; r < s; ++r) m01[r] = fmaxf(m01[r], m01[r + s]);
            float mx = fmaxf(m01[0], SWZ_XOR16(m01[0]));

            const float mnew  = fmaxf(mQ[qt], mx);
            const float alpha = __expf(mQ[qt] - mnew);
            mQ[qt] = mnew;
            alphaQ[qt] = alpha;

            // exponentials, pack to f16, store P row (two contiguous 16B chunks)
            U8h u0, u1;
            #pragma unroll
            for (int r = 0; r < 8; ++r) {
                u0.h[r] = (_Float16)__expf(ST[0][qt][r] - mnew);
                u1.h[r] = (_Float16)__expf(ST[1][qt][r] - mnew);
            }
            *(v8h*)&Ps[wave][qt * 16 + hl][rbase]      = u0.v;   // keys rbase..rbase+7
            *(v8h*)&Ps[wave][qt * 16 + hl][16 + rbase] = u1.v;   // keys 16+rbase..
        }

        // rescale accumulators: broadcast per-query alpha to C-layout rows
        #pragma unroll
        for (int qt = 0; qt < 2; ++qt)
            #pragma unroll
            for (int r = 0; r < 8; ++r) {
                const float a = bcast_lane(alphaQ[qt], idx4[r]);
                Oacc[qt][0][r] *= a;
                Oacc[qt][1][r] *= a;
                lacc[qt][r]    *= a;
            }

        // same-wave LDS RAW: drain DS pipe before re-reading P in A-layout
        asm volatile("s_wait_dscnt 0x0" ::: "memory");

        // A operand: P (M=query, Kdim=key) in A-layout
        U16 pa[2];
        #pragma unroll
        for (int qt = 0; qt < 2; ++qt) {
            const int d0k = hi ? 8 : 0;
            pa[qt].p[0] = *(const v8h*)&Ps[wave][qt * 16 + hl][d0k];
            pa[qt].p[1] = *(const v8h*)&Ps[wave][qt * 16 + hl][d0k + 16];
        }
        // B operand: V block (Kdim=key, N=d) from transposed V rows
        U16 vb[2];
        #pragma unroll
        for (int nt = 0; nt < 2; ++nt) {
            const int dcol = nt * 16 + hl;
            const int koff = k0 + (hi ? 8 : 0);
            vb[nt].p[0] = *(const v8h*)&Vt[dcol][koff];
            vb[nt].p[1] = *(const v8h*)&Vt[dcol][koff + 16];
        }
        #pragma unroll
        for (int qt = 0; qt < 2; ++qt) {
            #pragma unroll
            for (int nt = 0; nt < 2; ++nt)
                Oacc[qt][nt] = __builtin_amdgcn_wmma_f32_16x16x32_f16(
                    false, pa[qt].v, false, vb[nt].v, (short)0, Oacc[qt][nt], false, false);
            // softmax denominator via matrix unit: l += P * ones (row sums, C-layout)
            lacc[qt] = __builtin_amdgcn_wmma_f32_16x16x32_f16(
                false, pa[qt].v, false, ones.v, (short)0, lacc[qt], false, false);
        }
    }

    // ---- normalize (l already row-distributed) and store ----
    #pragma unroll
    for (int qt = 0; qt < 2; ++qt)
        #pragma unroll
        for (int r = 0; r < 8; ++r) {
            const float inv = 1.0f / lacc[qt][r];
            const int row   = q0 + qt * 16 + rbase + r;
            const size_t ob = goff(row);
            out[ob + hl]      = Oacc[qt][0][r] * inv;
            out[ob + 16 + hl] = Oacc[qt][1][r] * inv;
        }
}

extern "C" void kernel_launch(void* const* d_in, const int* in_sizes, int n_in,
                              void* d_out, int out_size, void* d_ws, size_t ws_size,
                              hipStream_t stream)
{
    const float* qkv  = (const float*)d_in[0];   // (3, 2, 65536, 192) fp32
    const float* bias = (const float*)d_in[1];   // (961, 6) fp32
    float* out = (float*)d_out;                  // (2, 256, 256, 192) fp32

    dim3 grid(3072);   // 2 * 256 windows * 6 heads
    dim3 block(256);   // 8 waves; each wave owns 32 query rows
    hipLaunchKernelGGL(cswin_attn_kernel, grid, block, 0, stream, qkv, bias, out);
}
